// BertBlock_77025943486655
// MI455X (gfx1250) — matrix operand order repaired and checked by reading, not compiled, not run on hardware
//
#include <hip/hip_runtime.h>
#include <hip/hip_bf16.h>
#include <math.h>
#include <stdint.h>

// ---------------- problem constants ----------------
static constexpr int Bc   = 4;
static constexpr int Sc   = 2048;
static constexpr int Hc   = 768;
static constexpr int NHc  = 12;
static constexpr int DHc  = 64;
static constexpr int FFc  = 3072;
static constexpr int BSc  = Bc * Sc;     // 8192 rows
static constexpr int QKVc = 3 * Hc;      // 2304
static constexpr float EPSc = 1e-5f;

typedef __bf16 bf16_t;
typedef __attribute__((ext_vector_type(16))) __bf16 v16bf;
typedef __attribute__((ext_vector_type(8)))  float  v8f;

// Toggle for the CDNA5 async global->LDS copy path (ASYNCcnt-tracked, no VGPR bounce).
#ifndef USE_ASYNC_COPY
#define USE_ASYNC_COPY 1
#endif

__device__ __forceinline__ void async_copy_b128(void* lds_dst, const void* gsrc) {
  // GLOBAL_LOAD_ASYNC_TO_LDS_B128: each lane copies 16 bytes from global memory to its LDS address.
  asm volatile("global_load_async_to_lds_b128 %0, %1, off"
               :
               : "v"((unsigned int)(uintptr_t)lds_dst), "v"(gsrc)
               : "memory");
}

__device__ __forceinline__ void async_wait0() {
#if __has_builtin(__builtin_amdgcn_s_wait_asynccnt)
  __builtin_amdgcn_s_wait_asynccnt(0);
#else
  asm volatile("s_wait_asynccnt 0x0" ::: "memory");
#endif
}

__device__ __forceinline__ v16bf frag2(uint4 lo, uint4 hi) {
  union { uint4 u[2]; v16bf v; } t;
  t.u[0] = lo; t.u[1] = hi;
  return t.v;
}

__device__ __forceinline__ v8f wmma_bf16(v16bf a, v16bf b, v8f c) {
  // D = A(16x32 bf16) * B(32x16 bf16) + C(16x16 f32)
  return __builtin_amdgcn_wmma_f32_16x16x32_bf16(
      false, a, false, b, (short)0, c, false, false);
}

__device__ __forceinline__ float gelu_exact(float x) {
  return 0.5f * x * (1.0f + erff(x * 0.70710678118654752440f));
}

// ---------------- fp32 -> bf16 conversion ----------------
__global__ __launch_bounds__(256)
void cvt_f32_bf16_kernel(const float* __restrict__ src, bf16_t* __restrict__ dst, int n) {
  int i = (blockIdx.x * 256 + threadIdx.x) * 4;
  if (i + 3 < n) {
    float4 v = *(const float4*)(src + i);
    dst[i + 0] = (bf16_t)v.x;
    dst[i + 1] = (bf16_t)v.y;
    dst[i + 2] = (bf16_t)v.z;
    dst[i + 3] = (bf16_t)v.w;
  } else {
    for (int k = i; k < n; ++k) dst[k] = (bf16_t)src[k];
  }
}

// ---------------- WMMA GEMM ----------------
// C[M,N] = A[M,K](bf16,row-major) * B[K,N](bf16,row-major) + bias
// MODE 0: store bf16.   MODE 1: GELU, store bf16.   MODE 2: +fp32 residual, store fp32.
static constexpr int BM = 128, BN = 128, BK = 32;
static constexpr int LDS_A = BK + 8;   // padded strides (bytes stay 16B-aligned)
static constexpr int LDS_B = BK + 8;

template <int MODE>
__global__ __launch_bounds__(256)
void gemm_bf16_kernel(const bf16_t* __restrict__ A, int lda,
                      const bf16_t* __restrict__ Bw, int ldb,
                      const float* __restrict__ bias,
                      const float* __restrict__ resF,
                      float* __restrict__ outF,
                      bf16_t* __restrict__ outB, int ldc,
                      int K) {
  __shared__ alignas(16) bf16_t As[BM * LDS_A];
  __shared__ alignas(16) bf16_t Bs[BN * LDS_B];   // stored transposed: [n][k]

  const int tid  = threadIdx.x;
  const int lane = tid & 31;
  const int wave = tid >> 5;         // 8 waves
  const int wx   = wave & 1;         // 2 waves along N
  const int wy   = wave >> 1;        // 4 waves along M
  const int bm0  = blockIdx.y * BM;
  const int bn0  = blockIdx.x * BN;
  const int wm0  = wy * 32;
  const int wn0  = wx * 64;
  const int col  = lane & 15;        // N index for B/C fragments, M row for A frag
  const int sub  = lane >> 4;        // half-wave selector

  v8f acc[2][4];
#pragma unroll
  for (int i = 0; i < 2; ++i)
#pragma unroll
    for (int j = 0; j < 4; ++j)
#pragma unroll
      for (int e = 0; e < 8; ++e) acc[i][j][e] = 0.0f;

  for (int k0 = 0; k0 < K; k0 += BK) {
    // A tile 128x32: 16B per thread, 2 passes (async direct-to-LDS, no VGPR bounce)
#pragma unroll
    for (int it = 0; it < 2; ++it) {
      const int idx = tid + it * 256;
      const int r = idx >> 2;
      const int c = (idx & 3) * 8;
#if USE_ASYNC_COPY
      async_copy_b128(As + r * LDS_A + c,
                      A + (size_t)(bm0 + r) * lda + k0 + c);
#else
      *(uint4*)(As + r * LDS_A + c) =
          *(const uint4*)(A + (size_t)(bm0 + r) * lda + k0 + c);
#endif
    }
    // B tile 32x128 -> transposed into Bs[n][k] (needs VGPR transpose)
#pragma unroll
    for (int it = 0; it < 2; ++it) {
      const int idx = tid + it * 256;
      const int r = idx >> 4;          // k 0..31
      const int c = (idx & 15) * 8;    // n 0..120
      union { uint4 u; bf16_t h8[8]; } tv;
      tv.u = *(const uint4*)(Bw + (size_t)(k0 + r) * ldb + bn0 + c);
#pragma unroll
      for (int e = 0; e < 8; ++e) Bs[(c + e) * LDS_B + r] = tv.h8[e];
    }
    if (k0 + BK < K) {  // -> global_prefetch_b8 for next K tile
      __builtin_prefetch(A + (size_t)(bm0 + (tid >> 2)) * lda + k0 + BK + (tid & 3) * 8, 0, 3);
      __builtin_prefetch(Bw + (size_t)(k0 + BK + (tid >> 4)) * ldb + bn0 + (tid & 15) * 8, 0, 3);
    }
#if USE_ASYNC_COPY
    async_wait0();
#endif
    __syncthreads();

    // A fragment: lane row = col, k chunks {sub*8.., sub*8+16..}
    v16bf afrag[2];
#pragma unroll
    for (int i = 0; i < 2; ++i) {
      const bf16_t* p = As + (wm0 + i * 16 + col) * LDS_A + sub * 8;
      afrag[i] = frag2(*(const uint4*)p, *(const uint4*)(p + 16));
    }
#pragma unroll
    for (int j = 0; j < 4; ++j) {
      // B fragment: lane column = col, k chunks {sub*8.., sub*8+16..} (transposed Bs)
      const bf16_t* p = Bs + (wn0 + j * 16 + col) * LDS_B + sub * 8;
      v16bf bfrag = frag2(*(const uint4*)p, *(const uint4*)(p + 16));
#pragma unroll
      for (int i = 0; i < 2; ++i)
        acc[i][j] = wmma_bf16(afrag[i], bfrag, acc[i][j]);
    }
    __syncthreads();
  }

  // epilogue: C/D layout row = r + 8*sub, col = lane&15
#pragma unroll
  for (int i = 0; i < 2; ++i) {
#pragma unroll
    for (int r = 0; r < 8; ++r) {
      const size_t gm = (size_t)(bm0 + wm0 + i * 16 + sub * 8 + r);
#pragma unroll
      for (int j = 0; j < 4; ++j) {
        const int gn = bn0 + wn0 + j * 16 + col;
        float v = acc[i][j][r] + bias[gn];
        if (MODE == 1) v = gelu_exact(v);
        if (MODE == 2) {
          v += resF[gm * ldc + gn];
          outF[gm * ldc + gn] = v;
        } else {
          outB[gm * ldc + gn] = (bf16_t)v;
        }
      }
    }
  }
}

// ---------------- Flash attention (bf16 WMMA, f32 online softmax) ----------------
// qkv: [B*S, 2304] bf16 with Q at +0, K at +768, V at +1536 (per-head slices of 64)
// out: attnb [B*S, 768] bf16  (heads concatenated)
static constexpr int ATT_LDS = 72;   // 64 + 8 pad, keeps 16B alignment (144B rows)

__global__ __launch_bounds__(128)
void attn_kernel(const bf16_t* __restrict__ qkv, bf16_t* __restrict__ attnb) {
  __shared__ alignas(16) bf16_t Ks[64 * ATT_LDS];        // [key][d]
  __shared__ alignas(16) bf16_t Vs[64 * ATT_LDS];        // [d][key] (transposed)
  __shared__ alignas(16) bf16_t Ps[4 * 16 * ATT_LDS];    // per-wave P strip [q][key]

  const int tid  = threadIdx.x;
  const int lane = tid & 31;
  const int wave = tid >> 5;     // 4 waves, 16 queries each
  const int col  = lane & 15;
  const int sub  = lane >> 4;

  int bid = blockIdx.x;
  const int qt = bid & 31;       // S/64 = 32 query tiles
  bid >>= 5;
  const int h = bid % NHc;
  const int b = bid / NHc;

  const size_t rstride = QKVc;
  const bf16_t* Qb = qkv + ((size_t)b * Sc) * rstride + (size_t)h * DHc;
  const bf16_t* Kb = Qb + Hc;
  const bf16_t* Vb = Qb + 2 * Hc;

  const int q0 = qt * 64 + wave * 16;

  // Q fragments (A-layout) straight from global: row = q0+col, k chunks per sub
  v16bf qf[2];
#pragma unroll
  for (int f = 0; f < 2; ++f) {
    const bf16_t* p = Qb + (size_t)(q0 + col) * rstride + f * 32 + sub * 8;
    qf[f] = frag2(*(const uint4*)p, *(const uint4*)(p + 16));
  }

  v8f o[4];
#pragma unroll
  for (int t = 0; t < 4; ++t)
#pragma unroll
    for (int e = 0; e < 8; ++e) o[t][e] = 0.0f;

  float mrow[8], lrow[8];
#pragma unroll
  for (int r = 0; r < 8; ++r) { mrow[r] = -1e30f; lrow[r] = 0.0f; }

  for (int kb = 0; kb < Sc; kb += 64) {
    // stage K (row-major, async direct-to-LDS) and V (transposed via VGPRs) tiles
#pragma unroll
    for (int it = 0; it < 4; ++it) {
      const int idx = tid + it * 128;
      const int kr = idx >> 3;        // key 0..63
      const int kc = (idx & 7) * 8;   // d   0..56
#if USE_ASYNC_COPY
      async_copy_b128(Ks + kr * ATT_LDS + kc,
                      Kb + (size_t)(kb + kr) * rstride + kc);
#else
      *(uint4*)(Ks + kr * ATT_LDS + kc) =
          *(const uint4*)(Kb + (size_t)(kb + kr) * rstride + kc);
#endif
      union { uint4 u; bf16_t h8[8]; } tv;
      tv.u = *(const uint4*)(Vb + (size_t)(kb + kr) * rstride + kc);
#pragma unroll
      for (int e = 0; e < 8; ++e) Vs[(kc + e) * ATT_LDS + kr] = tv.h8[e];
    }
#if USE_ASYNC_COPY
    async_wait0();
#endif
    __syncthreads();

    // S = Q * K^T / sqrt(DH): B-fragment column = key row of Ks (contiguous in d)
    v8f sacc[4];
#pragma unroll
    for (int nt = 0; nt < 4; ++nt) {
#pragma unroll
      for (int e = 0; e < 8; ++e) sacc[nt][e] = 0.0f;
#pragma unroll
      for (int f = 0; f < 2; ++f) {
        const bf16_t* p = Ks + (nt * 16 + col) * ATT_LDS + f * 32 + sub * 8;
        v16bf kf = frag2(*(const uint4*)p, *(const uint4*)(p + 16));
        sacc[nt] = wmma_bf16(qf[f], kf, sacc[nt]);
      }
#pragma unroll
      for (int e = 0; e < 8; ++e) sacc[nt][e] *= 0.125f;  // 1/sqrt(64)
    }

    // online softmax: row = r + 8*sub lives across 16 lanes of a half-wave
    float mnew[8], alpha[8];
#pragma unroll
    for (int r = 0; r < 8; ++r) {
      float mx = sacc[0][r];
#pragma unroll
      for (int nt = 1; nt < 4; ++nt) mx = fmaxf(mx, sacc[nt][r]);
#pragma unroll
      for (int d = 1; d <= 8; d <<= 1) mx = fmaxf(mx, __shfl_xor(mx, d, 32));
      mnew[r]  = fmaxf(mrow[r], mx);
      alpha[r] = __expf(mrow[r] - mnew[r]);
      mrow[r]  = mnew[r];
    }
#pragma unroll
    for (int r = 0; r < 8; ++r) {
      float s = 0.0f;
#pragma unroll
      for (int nt = 0; nt < 4; ++nt) {
        float e0 = __expf(sacc[nt][r] - mnew[r]);
        sacc[nt][r] = e0;
        s += e0;
      }
#pragma unroll
      for (int d = 1; d <= 8; d <<= 1) s += __shfl_xor(s, d, 32);
      lrow[r] = lrow[r] * alpha[r] + s;
    }
#pragma unroll
    for (int t = 0; t < 4; ++t)
#pragma unroll
      for (int r = 0; r < 8; ++r) o[t][r] *= alpha[r];

    // bounce P through per-wave LDS strip to re-read in A-layout (same-wave LDS is in-order)
    bf16_t* Pw = Ps + wave * 16 * ATT_LDS;
#pragma unroll
    for (int nt = 0; nt < 4; ++nt)
#pragma unroll
      for (int r = 0; r < 8; ++r)
        Pw[(sub * 8 + r) * ATT_LDS + nt * 16 + col] = (bf16_t)sacc[nt][r];

    // O += P * V  (V transposed in LDS -> contiguous key per d column)
#pragma unroll
    for (int f = 0; f < 2; ++f) {
      const bf16_t* pp = Pw + col * ATT_LDS + f * 32 + sub * 8;
      v16bf pf = frag2(*(const uint4*)pp, *(const uint4*)(pp + 16));
#pragma unroll
      for (int t = 0; t < 4; ++t) {
        const bf16_t* vp = Vs + (t * 16 + col) * ATT_LDS + f * 32 + sub * 8;
        v16bf vf = frag2(*(const uint4*)vp, *(const uint4*)(vp + 16));
        o[t] = wmma_bf16(pf, vf, o[t]);
      }
    }
    __syncthreads();
  }

#pragma unroll
  for (int r = 0; r < 8; ++r) {
    const float inv = 1.0f / lrow[r];
    const size_t gm = (size_t)b * Sc + q0 + sub * 8 + r;
#pragma unroll
    for (int t = 0; t < 4; ++t)
      attnb[gm * Hc + h * DHc + t * 16 + col] = (bf16_t)(o[t][r] * inv);
  }
}

// ---------------- LayerNorm (row = 768 fp32) ----------------
__global__ __launch_bounds__(256)
void layernorm_kernel(const float* __restrict__ in,
                      const float* __restrict__ w,
                      const float* __restrict__ bb,
                      float* __restrict__ outF,
                      bf16_t* __restrict__ outB) {
  __shared__ float redS[8], redSS[8];
  __shared__ float s_mean, s_rstd;
  const int rowid = blockIdx.x;
  const float* x = in + (size_t)rowid * Hc;
  float s = 0.f, ss = 0.f;
  for (int c = threadIdx.x; c < Hc; c += 256) { float v = x[c]; s += v; ss += v * v; }
#pragma unroll
  for (int d = 16; d >= 1; d >>= 1) { s += __shfl_xor(s, d, 32); ss += __shfl_xor(ss, d, 32); }
  if ((threadIdx.x & 31) == 0) { redS[threadIdx.x >> 5] = s; redSS[threadIdx.x >> 5] = ss; }
  __syncthreads();
  if (threadIdx.x == 0) {
    float S = 0.f, SS = 0.f;
    for (int i = 0; i < 8; ++i) { S += redS[i]; SS += redSS[i]; }
    const float mean = S / (float)Hc;
    s_mean = mean;
    s_rstd = rsqrtf(fmaxf(SS / (float)Hc - mean * mean, 0.0f) + EPSc);
  }
  __syncthreads();
  const float mean = s_mean, rstd = s_rstd;
  for (int c = threadIdx.x; c < Hc; c += 256) {
    float v = (x[c] - mean) * rstd * w[c] + bb[c];
    if (outF) outF[(size_t)rowid * Hc + c] = v;
    if (outB) outB[(size_t)rowid * Hc + c] = (bf16_t)v;
  }
}

// ---------------- host orchestration ----------------
extern "C" void kernel_launch(void* const* d_in, const int* in_sizes, int n_in,
                              void* d_out, int out_size, void* d_ws, size_t ws_size,
                              hipStream_t stream) {
  (void)in_sizes; (void)n_in; (void)out_size; (void)ws_size;
  const float* x    = (const float*)d_in[0];
  const float* Wq   = (const float*)d_in[1];
  const float* bq   = (const float*)d_in[2];
  const float* Wk   = (const float*)d_in[3];
  const float* bk   = (const float*)d_in[4];
  const float* Wv   = (const float*)d_in[5];
  const float* bv   = (const float*)d_in[6];
  const float* Wo   = (const float*)d_in[7];
  const float* bo   = (const float*)d_in[8];
  const float* W1   = (const float*)d_in[9];
  const float* b1   = (const float*)d_in[10];
  const float* W2   = (const float*)d_in[11];
  const float* b2   = (const float*)d_in[12];
  const float* ln1w = (const float*)d_in[13];
  const float* ln1b = (const float*)d_in[14];
  const float* ln2w = (const float*)d_in[15];
  const float* ln2b = (const float*)d_in[16];
  float* out = (float*)d_out;

  char* ws = (char*)d_ws;
  size_t off = 0;
  auto take = [&](size_t bytes) -> char* {
    char* p = ws + off;
    off += (bytes + 255) & ~(size_t)255;
    return p;
  };

  bf16_t* xb   = (bf16_t*)take((size_t)BSc * Hc * 2);
  bf16_t* wqb  = (bf16_t*)take((size_t)Hc * Hc * 2);
  bf16_t* wkb  = (bf16_t*)take((size_t)Hc * Hc * 2);
  bf16_t* wvb  = (bf16_t*)take((size_t)Hc * Hc * 2);
  bf16_t* wob  = (bf16_t*)take((size_t)Hc * Hc * 2);
  bf16_t* w1b  = (bf16_t*)take((size_t)Hc * FFc * 2);
  bf16_t* w2b  = (bf16_t*)take((size_t)FFc * Hc * 2);
  bf16_t* qkvb = (bf16_t*)take((size_t)BSc * QKVc * 2);   // 37.75 MB
  bf16_t* attb = (bf16_t*)take((size_t)BSc * Hc * 2);     // 12.58 MB (contiguous after qkvb)
  float*  y1   = (float*)take((size_t)BSc * Hc * 4);
  float*  x1f  = (float*)take((size_t)BSc * Hc * 4);
  bf16_t* x1b  = (bf16_t*)take((size_t)BSc * Hc * 2);
  bf16_t* hb   = qkvb;   // FFN hidden (50.33 MB) reuses dead qkvb+attb (37.75+12.58 MB)
  float*  y2   = y1;     // pre-LN2 sum reuses dead y1

  auto cvt = [&](const float* s, bf16_t* d, int n) {
    cvt_f32_bf16_kernel<<<dim3((unsigned)((n / 4 + 255) / 256)), dim3(256), 0, stream>>>(s, d, n);
  };
  cvt(x,  xb,  BSc * Hc);
  cvt(Wq, wqb, Hc * Hc);
  cvt(Wk, wkb, Hc * Hc);
  cvt(Wv, wvb, Hc * Hc);
  cvt(Wo, wob, Hc * Hc);
  cvt(W1, w1b, Hc * FFc);
  cvt(W2, w2b, FFc * Hc);

  const dim3 blk(256);
  const dim3 gH(Hc / BN, BSc / BM);    // 6 x 64
  const dim3 gF(FFc / BN, BSc / BM);   // 24 x 64

  // QKV projections into strided [BS, 2304] buffer
  gemm_bf16_kernel<0><<<gH, blk, 0, stream>>>(xb, Hc, wqb, Hc, bq, nullptr, nullptr,
                                              qkvb + 0 * Hc, QKVc, Hc);
  gemm_bf16_kernel<0><<<gH, blk, 0, stream>>>(xb, Hc, wkb, Hc, bk, nullptr, nullptr,
                                              qkvb + 1 * Hc, QKVc, Hc);
  gemm_bf16_kernel<0><<<gH, blk, 0, stream>>>(xb, Hc, wvb, Hc, bv, nullptr, nullptr,
                                              qkvb + 2 * Hc, QKVc, Hc);

  // Flash attention: B*NH*(S/64) = 1536 blocks of 128 threads
  attn_kernel<<<dim3(Bc * NHc * (Sc / 64)), dim3(128), 0, stream>>>(qkvb, attb);

  // O-projection + residual(x) -> y1 (fp32)
  gemm_bf16_kernel<2><<<gH, blk, 0, stream>>>(attb, Hc, wob, Hc, bo, x, y1, nullptr, Hc, Hc);

  // LN1 -> x1 (fp32 residual stream + bf16 GEMM input)
  layernorm_kernel<<<dim3(BSc), dim3(256), 0, stream>>>(y1, ln1w, ln1b, x1f, x1b);

  // FFN up + exact GELU -> hb (bf16)
  gemm_bf16_kernel<1><<<gF, blk, 0, stream>>>(x1b, Hc, w1b, FFc, b1, nullptr, nullptr,
                                              hb, FFc, Hc);

  // FFN down + residual(x1) -> y2 (fp32)
  gemm_bf16_kernel<2><<<gH, blk, 0, stream>>>(hb, FFc, w2b, Hc, b2, x1f, y2, nullptr, Hc, FFc);

  // LN2 -> output (fp32)
  layernorm_kernel<<<dim3(BSc), dim3(256), 0, stream>>>(y2, ln2w, ln2b, out, nullptr);
}
